// CharTransformer_16252156248283
// MI455X (gfx1250) — matrix operand order repaired and checked by reading, not compiled, not run on hardware
//
#include <hip/hip_runtime.h>
#include <hip/hip_bf16.h>

// ---------------------------------------------------------------------------
// CharTransformer forward for MI455X (gfx1250).
// bf16 WMMA for all matmuls, TDM (tensor_load_to_lds) for GEMM A-tile staging,
// double-buffered LDS, flash attention with online softmax.
// ---------------------------------------------------------------------------

typedef __bf16 bf16_t;
typedef bf16_t v16bf __attribute__((ext_vector_type(16)));
typedef float  v8f   __attribute__((ext_vector_type(8)));
typedef unsigned int u32x4 __attribute__((ext_vector_type(4)));
typedef int          i32x4 __attribute__((ext_vector_type(4)));
typedef int          i32x8 __attribute__((ext_vector_type(8)));

union V16U { v16bf v; uint4 u[2]; bf16_t e[16]; };
static_assert(sizeof(v16bf) == 32, "v16bf must be 32 bytes");

constexpr int Bc = 2, T = 2048, E = 1024, NL = 6, H = 16, Dh = 64, FF = 4096, NV = 256;
constexpr int Mrows = Bc * T;   // 4096

#if __has_builtin(__builtin_amdgcn_tensor_load_to_lds) && \
    __has_builtin(__builtin_amdgcn_s_wait_tensorcnt)
#define TDM_OK 1
#else
#define TDM_OK 0
#endif

__device__ __forceinline__ v8f wmma_bf16(v16bf a, v16bf b, v8f c) {
  // D = A(16x32) * B(32x16) + C, f32 accumulate
  return __builtin_amdgcn_wmma_f32_16x16x32_bf16(false, a, false, b, (short)0, c,
                                                 false, false);
}

// ---------------------------------------------------------------------------
// Embedding: x[b,t,:] = embed[idx[b,t],:] + pos_embed[t,:]
// ---------------------------------------------------------------------------
__global__ __launch_bounds__(256) void k_embed(const int* __restrict__ idx,
                                               const float* __restrict__ emb,
                                               const float* __restrict__ pos,
                                               float* __restrict__ x) {
  const int m = blockIdx.x;
  const int t = m & (T - 1);
  const int tok = idx[m];
  for (int e = threadIdx.x; e < E; e += 256)
    x[(size_t)m * E + e] = emb[(size_t)tok * E + e] + pos[(size_t)t * E + e];
}

// ---------------------------------------------------------------------------
// LayerNorm (f32 in, bf16 out for GEMM A operand)
// ---------------------------------------------------------------------------
__global__ __launch_bounds__(256) void k_ln(const float* __restrict__ x,
                                            const float* __restrict__ gam,
                                            const float* __restrict__ bet,
                                            bf16_t* __restrict__ xn) {
  const int m = blockIdx.x;
  __shared__ float red0[8], red1[8];
  const float* xr = x + (size_t)m * E;
  float s = 0.f, s2 = 0.f;
  for (int e = threadIdx.x; e < E; e += 256) {
    float v = xr[e];
    s += v; s2 += v * v;
  }
#pragma unroll
  for (int o = 16; o >= 1; o >>= 1) {
    s  += __shfl_xor(s,  o, 32);
    s2 += __shfl_xor(s2, o, 32);
  }
  const int wv = threadIdx.x >> 5, lane = threadIdx.x & 31;
  if (lane == 0) { red0[wv] = s; red1[wv] = s2; }
  __syncthreads();
  s = 0.f; s2 = 0.f;
#pragma unroll
  for (int i = 0; i < 8; ++i) { s += red0[i]; s2 += red1[i]; }
  const float mean = s * (1.f / E);
  const float var  = s2 * (1.f / E) - mean * mean;
  const float rstd = rsqrtf(var + 1e-5f);
  for (int e = threadIdx.x; e < E; e += 256) {
    float v = (xr[e] - mean) * rstd * gam[e] + bet[e];
    xn[(size_t)m * E + e] = (bf16_t)v;
  }
}

// ---------------------------------------------------------------------------
// bf16 WMMA GEMM: C(M x N) = A(M x K, bf16 row-major) * Bw(K x N, f32)
// Block tile 128x128, 256 threads = 8 waves, each wave 64x32 (8 WMMA tiles).
// A tiles: TDM DMA (tensor_load_to_lds) into row-major LDS, double-buffered.
// B tiles: f32 -> bf16 converted into fragment-order LDS by all threads.
//   A lane (m = lane&15, g = lane>>4): K run  [16g..16g+15]
//   B lane (n = lane&15, g = lane>>4): K run  [16g..16g+15]
// Epilogues:
//   0: Q  -> bf16 [b,h,t,d], scaled by D^-0.5
//   1: K  -> bf16 [b,h,t,d]
//   2: V  -> bf16 [b,h,d,t]   (transposed: attention B-fragments contiguous)
//   3: residual: xio += acc + bias   (f32)
//   4: relu(acc + bias) -> bf16
//   5: acc + bias -> f32 (logits)
// ---------------------------------------------------------------------------
template <int EPI>
__global__ __launch_bounds__(256) void k_gemm(const bf16_t* __restrict__ A,
                                              const float* __restrict__ Bw,
                                              const float* __restrict__ bias,
                                              float* __restrict__ xio,
                                              bf16_t* __restrict__ obf,
                                              float* __restrict__ of32,
                                              int Ndim, int Kdim, int headed) {
  __shared__ bf16_t lA[2][128 * 32];   // row-major A tile (TDM destination)
  __shared__ bf16_t lB[2][8 * 512];    // fragment-order B tile
  const int tid  = threadIdx.x;
  const int lane = tid & 31, wv = tid >> 5;
  const int g = lane >> 4, ml = lane & 15;
  const int m_base = blockIdx.y * 128, n_base = blockIdx.x * 128;
  const int wm = wv & 1, wn = wv >> 1;   // wave -> (2 M) x (4 N) grid

  const v8f vzero = {0.f, 0.f, 0.f, 0.f, 0.f, 0.f, 0.f, 0.f};
  v8f acc[4][2];
#pragma unroll
  for (int i = 0; i < 4; ++i)
#pragma unroll
    for (int j = 0; j < 2; ++j) acc[i][j] = vzero;

  // ---- A tile load: TDM DMA (wave 0 issues) or cooperative-copy fallback ----
  auto loadA = [&](int kb, int buf) {
#if TDM_OK
    if (wv == 0) {
      const unsigned long long ga =
          (unsigned long long)(const void*)(A + (size_t)m_base * Kdim + kb);
      const unsigned lds_addr =
          (unsigned)(unsigned long long)(const void*)&lA[buf][0];
      // D# group 0: count=1 | lds_addr | global_addr(57b) | type=2
      u32x4 g0;
      g0[0] = 1u;
      g0[1] = lds_addr;
      g0[2] = (unsigned)ga;
      g0[3] = (unsigned)((ga >> 32) & 0x01FFFFFFull) | 0x80000000u;
      // D# group 1: data_size=2B, tensor 2D (Kdim x Mrows), tile 32 x 128
      i32x8 g1;
      g1[0] = 0x10000;                                          // data_size=1 (2B)
      g1[1] = (int)(((unsigned)Kdim & 0xFFFFu) << 16);          // tensor_dim0 lo
      g1[2] = (int)(((unsigned)Kdim >> 16) |
                    (((unsigned)Mrows & 0xFFFFu) << 16));       // dim0 hi | dim1 lo
      g1[3] = (int)(((unsigned)Mrows >> 16) | (32u << 16));     // dim1 hi | tile_dim0
      g1[4] = 128;                                              // tile_dim1 (tile_dim2=0)
      g1[5] = Kdim;                                             // tensor_dim0_stride
      g1[6] = 0;
      g1[7] = 0;
      i32x4 g2 = {0, 0, 0, 0};
      i32x4 g3 = {0, 0, 0, 0};
#if defined(__clang_major__) && (__clang_major__ >= 23)
      i32x8 g4 = {0, 0, 0, 0, 0, 0, 0, 0};
      __builtin_amdgcn_tensor_load_to_lds(g0, g1, g2, g3, g4, 0);
#else
      __builtin_amdgcn_tensor_load_to_lds(g0, g1, g2, g3, 0);
#endif
    }
#else
    const int row = tid >> 1, half = tid & 1;
    const bf16_t* src = A + (size_t)(m_base + row) * Kdim + kb + half * 16;
    uint4 c0 = *reinterpret_cast<const uint4*>(src);
    uint4 c1 = *reinterpret_cast<const uint4*>(src + 8);
    uint4* dst = reinterpret_cast<uint4*>(&lA[buf][row * 32 + half * 16]);
    dst[0] = c0; dst[1] = c1;
#endif
  };
  auto waitA = [&]() {
#if TDM_OK
    if (wv == 0) __builtin_amdgcn_s_wait_tensorcnt(0);
#endif
  };
  // ---- B tile: f32 -> bf16 into fragment order ----
  const int b_col = n_base + wv * 16 + ml;
  auto convB = [&](int kb, int buf) {
    V16U tv;
#pragma unroll
    for (int e2 = 0; e2 < 16; ++e2) {
      const int k = kb + 16 * g + e2;
      size_t off;
      if (headed)  // (H,E,D) head-blocked weight: col = h*64 + d
        off = (size_t)(b_col >> 6) * ((size_t)Kdim * 64) + (size_t)k * 64 +
              (b_col & 63);
      else
        off = (size_t)k * Ndim + b_col;
      tv.e[e2] = (bf16_t)Bw[off];
    }
    *reinterpret_cast<v16bf*>(&lB[buf][wv * 512 + lane * 16]) = tv.v;
  };

  // ---- prologue: stage tile 0 ----
  loadA(0, 0);
  convB(0, 0);
  waitA();
  __syncthreads();

  // ---- main loop: compute buf `cur` while staging buf `nxt` ----
  for (int kb = 0; kb < Kdim; kb += 32) {
    const int cur = (kb >> 5) & 1, nxt = cur ^ 1;
    const bool more = (kb + 32) < Kdim;
    if (more) {
      loadA(kb + 32, nxt);
      convB(kb + 32, nxt);
    }
    v16bf af[4], bfr[2];
#pragma unroll
    for (int i = 0; i < 4; ++i) {
      const bf16_t* ar = &lA[cur][((wm * 4 + i) * 16 + ml) * 32];
      V16U tv;
      tv.u[0] = *reinterpret_cast<const uint4*>(ar + 16 * g);
      tv.u[1] = *reinterpret_cast<const uint4*>(ar + 16 * g + 8);
      af[i] = tv.v;
    }
#pragma unroll
    for (int j = 0; j < 2; ++j)
      bfr[j] = *reinterpret_cast<const v16bf*>(&lB[cur][(wn * 2 + j) * 512 + lane * 16]);
#pragma unroll
    for (int i = 0; i < 4; ++i)
#pragma unroll
      for (int j = 0; j < 2; ++j)
        acc[i][j] = wmma_bf16(af[i], bfr[j], acc[i][j]);
    if (more) waitA();
    __syncthreads();
  }

  // ---- epilogue (C layout: n = lane&15, m = v + 8*(lane>>4)) ----
#pragma unroll
  for (int i = 0; i < 4; ++i) {
#pragma unroll
    for (int j = 0; j < 2; ++j) {
      const int c = n_base + (wn * 2 + j) * 16 + ml;
#pragma unroll
      for (int v = 0; v < 8; ++v) {
        const int r = m_base + (wm * 4 + i) * 16 + v + 8 * g;
        float val = acc[i][j][v];
        if constexpr (EPI == 0) {
          const int b = r >> 11, t = r & (T - 1), h = c >> 6, d = c & 63;
          obf[(((size_t)(b * H + h)) * T + t) * Dh + d] = (bf16_t)(val * 0.125f);
        } else if constexpr (EPI == 1) {
          const int b = r >> 11, t = r & (T - 1), h = c >> 6, d = c & 63;
          obf[(((size_t)(b * H + h)) * T + t) * Dh + d] = (bf16_t)val;
        } else if constexpr (EPI == 2) {
          const int b = r >> 11, t = r & (T - 1), h = c >> 6, d = c & 63;
          obf[(((size_t)(b * H + h)) * Dh + d) * T + t] = (bf16_t)val;
        } else if constexpr (EPI == 3) {
          const size_t o = (size_t)r * Ndim + c;
          xio[o] = xio[o] + val + bias[c];
        } else if constexpr (EPI == 4) {
          const float z = val + bias[c];
          obf[(size_t)r * Ndim + c] = (bf16_t)(z > 0.f ? z : 0.f);
        } else {
          of32[(size_t)r * Ndim + c] = val + bias[c];
        }
      }
    }
  }
}

// ---------------------------------------------------------------------------
// Flash attention: one wave per (b,h, 16-row strip). Causal, online softmax.
// q: [b,h,t,d] bf16 (pre-scaled), k: [b,h,t,d] bf16, v: [b,h,d,t] bf16.
// Output o: row-major (Bc*T) x E bf16 (col = h*64 + d).
// ---------------------------------------------------------------------------
__global__ __launch_bounds__(256) void k_attn(const bf16_t* __restrict__ qb,
                                              const bf16_t* __restrict__ kkb,
                                              const bf16_t* __restrict__ vt,
                                              bf16_t* __restrict__ ob) {
  __shared__ bf16_t pbuf[8][16 * 32];    // per-wave P staging (C-layout -> A-frag)
  const int lane = threadIdx.x & 31, wv = threadIdx.x >> 5;
  const int gid = blockIdx.x * 8 + wv;   // 4096 waves total
  const int ttile = gid & (T / 16 - 1);
  const int bh = gid >> 7;               // b*H + h
  const int t0 = ttile * 16;
  const int g = lane >> 4, ml = lane & 15;

  // Q fragments: rows t0..t0+15, d split 0..31 / 32..63
  v16bf qf[2];
  {
    const bf16_t* qr = qb + ((size_t)bh * T + t0 + ml) * Dh;
#pragma unroll
    for (int h2 = 0; h2 < 2; ++h2) {
      V16U tv;
      tv.u[0] = *reinterpret_cast<const uint4*>(qr + h2 * 32 + 16 * g);
      tv.u[1] = *reinterpret_cast<const uint4*>(qr + h2 * 32 + 16 * g + 8);
      qf[h2] = tv.v;
    }
  }

  const v8f vzero = {0.f, 0.f, 0.f, 0.f, 0.f, 0.f, 0.f, 0.f};
  float mrow[8], lrow[8];
  v8f oacc[4];
#pragma unroll
  for (int v = 0; v < 8; ++v) { mrow[v] = -1e30f; lrow[v] = 0.f; }
#pragma unroll
  for (int dt = 0; dt < 4; ++dt) oacc[dt] = vzero;

  const int nk = t0 + 16;                // number of visible keys
  for (int s0 = 0; s0 < nk; s0 += 32) {
    // ---- scores S(16x32) = Q(16x64) * K^T ----
    v8f sacc[2];
#pragma unroll
    for (int stile = 0; stile < 2; ++stile) {
      const bf16_t* kr = kkb + ((size_t)bh * T + s0 + stile * 16 + ml) * Dh;
      v16bf kf0 = *reinterpret_cast<const v16bf*>(kr + 16 * g);
      v16bf kf1 = *reinterpret_cast<const v16bf*>(kr + 32 + 16 * g);
      v8f tacc = wmma_bf16(qf[0], kf0, vzero);
      sacc[stile] = wmma_bf16(qf[1], kf1, tacc);
    }
    // ---- causal mask + online softmax (row = v + 8g, half-wave reductions) --
#pragma unroll
    for (int v = 0; v < 8; ++v) {
      const int trow = t0 + v + 8 * g;
      float x0 = sacc[0][v]; if (s0 + ml > trow)      x0 = -1e30f;
      float x1 = sacc[1][v]; if (s0 + 16 + ml > trow) x1 = -1e30f;
      float mx = fmaxf(x0, x1);
#pragma unroll
      for (int o = 8; o >= 1; o >>= 1) mx = fmaxf(mx, __shfl_xor(mx, o, 32));
      const float mn = fmaxf(mrow[v], mx);
      const float al = __expf(mrow[v] - mn);
      mrow[v] = mn;
      const float p0 = __expf(x0 - mn);
      const float p1 = __expf(x1 - mn);
      float rs = p0 + p1;
#pragma unroll
      for (int o = 8; o >= 1; o >>= 1) rs += __shfl_xor(rs, o, 32);
      lrow[v] = lrow[v] * al + rs;
#pragma unroll
      for (int dt = 0; dt < 4; ++dt) oacc[dt][v] = oacc[dt][v] * al;
      pbuf[wv][(v + 8 * g) * 32 + ml]      = (bf16_t)p0;
      pbuf[wv][(v + 8 * g) * 32 + 16 + ml] = (bf16_t)p1;
    }
    asm volatile("s_wait_dscnt 0" ::: "memory");   // wave-local LDS RAW fence
    // ---- reload P as A-fragment ----
    v16bf pf;
    {
      V16U tv;
      const bf16_t* pr = &pbuf[wv][ml * 32];
      tv.u[0] = *reinterpret_cast<const uint4*>(pr + 16 * g);
      tv.u[1] = *reinterpret_cast<const uint4*>(pr + 16 * g + 8);
      pf = tv.v;
    }
    // ---- O += P(16x32) * V(32x64) ----
#pragma unroll
    for (int dt = 0; dt < 4; ++dt) {
      const bf16_t* vr = vt + ((size_t)bh * Dh + dt * 16 + ml) * T + s0 + 16 * g;
      v16bf vf = *reinterpret_cast<const v16bf*>(vr);
      oacc[dt] = wmma_bf16(pf, vf, oacc[dt]);
    }
  }

  const int b = bh >> 4, h = bh & 15;
#pragma unroll
  for (int dt = 0; dt < 4; ++dt)
#pragma unroll
    for (int v = 0; v < 8; ++v) {
      const int trow = t0 + v + 8 * g;
      const float o = oacc[dt][v] / lrow[v];
      ob[((size_t)b * T + trow) * E + h * Dh + dt * 16 + ml] = (bf16_t)o;
    }
}

// ---------------------------------------------------------------------------
// Host-side orchestration
// ---------------------------------------------------------------------------
extern "C" void kernel_launch(void* const* d_in, const int* in_sizes, int n_in,
                              void* d_out, int out_size, void* d_ws, size_t ws_size,
                              hipStream_t stream) {
  const int*   idx  = (const int*)d_in[0];
  const float* emb  = (const float*)d_in[1];
  const float* pos  = (const float*)d_in[2];
  const float* Wq   = (const float*)d_in[3];
  const float* Wk   = (const float*)d_in[4];
  const float* Wv   = (const float*)d_in[5];
  const float* Wo   = (const float*)d_in[6];
  const float* bo   = (const float*)d_in[7];
  const float* W1   = (const float*)d_in[8];
  const float* b1   = (const float*)d_in[9];
  const float* W2   = (const float*)d_in[10];
  const float* b2   = (const float*)d_in[11];
  const float* ln1g = (const float*)d_in[12];
  const float* ln1b = (const float*)d_in[13];
  const float* ln2g = (const float*)d_in[14];
  const float* ln2b = (const float*)d_in[15];
  const float* lnfg = (const float*)d_in[16];
  const float* lnfb = (const float*)d_in[17];
  const float* lmW  = (const float*)d_in[18];
  const float* lmb  = (const float*)d_in[19];
  float* out = (float*)d_out;

  // Workspace layout (bytes)
  char* ws = (char*)d_ws;
  float*  x   = (float*)(ws + 0);          // 4096*1024 f32  = 16 MB
  bf16_t* xn  = (bf16_t*)(ws + 16777216);  // 4096*1024 bf16 =  8 MB
  bf16_t* qb  = (bf16_t*)(ws + 25165824);  //  8 MB
  bf16_t* kb  = (bf16_t*)(ws + 33554432);  //  8 MB
  bf16_t* vtb = (bf16_t*)(ws + 41943040);  //  8 MB (transposed)
  bf16_t* obf = (bf16_t*)(ws + 50331648);  //  8 MB
  bf16_t* hb  = (bf16_t*)(ws + 58720256);  // 4096*4096 bf16 = 32 MB  -> 88 MB total

  const dim3 blk(256);
  const dim3 gE(Mrows);                       // embed / LN: one block per row
  const dim3 gProj(E / 128, Mrows / 128);     // N=1024
  const dim3 gF1(FF / 128, Mrows / 128);      // N=4096
  const dim3 gLM(NV / 128, Mrows / 128);      // N=256
  const dim3 gAttn(Mrows * H / (16 * 8));     // 512 blocks * 8 waves

  k_embed<<<gE, blk, 0, stream>>>(idx, emb, pos, x);

  for (int l = 0; l < NL; ++l) {
    const size_t wqkv = (size_t)l * H * E * Dh;
    k_ln<<<gE, blk, 0, stream>>>(x, ln1g + l * E, ln1b + l * E, xn);
    k_gemm<0><<<gProj, blk, 0, stream>>>(xn, Wq + wqkv, nullptr, nullptr, qb,  nullptr, E, E, 1);
    k_gemm<1><<<gProj, blk, 0, stream>>>(xn, Wk + wqkv, nullptr, nullptr, kb,  nullptr, E, E, 1);
    k_gemm<2><<<gProj, blk, 0, stream>>>(xn, Wv + wqkv, nullptr, nullptr, vtb, nullptr, E, E, 1);
    k_attn<<<gAttn, blk, 0, stream>>>(qb, kb, vtb, obf);
    k_gemm<3><<<gProj, blk, 0, stream>>>(obf, Wo + (size_t)l * E * E, bo + l * E, x,
                                         nullptr, nullptr, E, E, 0);
    k_ln<<<gE, blk, 0, stream>>>(x, ln2g + l * E, ln2b + l * E, xn);
    k_gemm<4><<<gF1, blk, 0, stream>>>(xn, W1 + (size_t)l * E * FF, b1 + l * FF,
                                       nullptr, hb, nullptr, FF, E, 0);
    k_gemm<3><<<gProj, blk, 0, stream>>>(hb, W2 + (size_t)l * FF * E, b2 + l * E, x,
                                         nullptr, nullptr, E, FF, 0);
  }

  k_ln<<<gE, blk, 0, stream>>>(x, lnfg, lnfb, xn);
  k_gemm<5><<<gLM, blk, 0, stream>>>(xn, lmW, lmb, nullptr, nullptr, out, NV, E, 0);
}
